// MultiHeadAttention_27092653703172
// MI455X (gfx1250) — matrix operand order repaired and checked by reading, not compiled, not run on hardware
//
#include <hip/hip_runtime.h>

// ---------------------------------------------------------------------------
// MHA forward for MI455X (gfx1250, wave32, WMMA).
// B=2, SQ=SK=2048, D_MODEL=1024, H=16, DEPTH=64.
// d_out = [ out (2*2048*1024 f32) | attn (2*16*2048*2048 f32) ]
// d_ws  = [ qh | kh | vh (each [b][h][s][64] f16) | o_concat [b][s][1024] f16 ]
// ---------------------------------------------------------------------------

#define DEV static __device__ __forceinline__

typedef __attribute__((ext_vector_type(16))) _Float16 v16h;
typedef __attribute__((ext_vector_type(8)))  _Float16 v8h;
typedef __attribute__((ext_vector_type(8)))  float    v8f;
typedef __attribute__((ext_vector_type(4)))  float    f32x4;
typedef int v4i __attribute__((vector_size(16)));   // matches builtin V4i

constexpr int SEQ   = 2048;          // SQ == SK
constexpr int DM    = 1024;          // d_model
constexpr int NH    = 16;
constexpr int DH    = 64;            // depth per head
constexpr int MROWS = 2 * SEQ;       // B*S = 4096

// ---- gfx1250 async global->LDS copy (ASYNCcnt path), with safe fallback ----
#if defined(__has_builtin)
#if __has_builtin(__builtin_amdgcn_global_load_async_to_lds_b128) && \
    __has_builtin(__builtin_amdgcn_s_wait_asynccnt)
#define USE_ASYNC_LDS 1
#endif
#endif

// Copy 16 bytes (8 f16) from global to LDS, per lane.
DEV void cp16_g2l(_Float16* lds, const _Float16* g) {
#if defined(USE_ASYNC_LDS)
  __builtin_amdgcn_global_load_async_to_lds_b128(
      (__attribute__((address_space(1))) v4i*)g,
      (__attribute__((address_space(3))) v4i*)lds, 0, 0);
#else
  *(v8h*)lds = *(const v8h*)g;
#endif
}

DEV void async_copy_fence() {
#if defined(USE_ASYNC_LDS)
  __builtin_amdgcn_s_wait_asynccnt(0);
#endif
}

DEV v8f wmma16(v16h a, v16h b, v8f c) {
  // v_wmma_f32_16x16x32_f16 : D = A(16x32) * B(32x16) + C, fp32 accum
  return __builtin_amdgcn_wmma_f32_16x16x32_f16(false, a, false, b,
                                                (short)0, c, false, false);
}

// A fragment (16x32 f16, MxK). ISA layout: lane = hi*16 + m,
// element e -> k = (e>>3)*16 + hi*8 + (e&7).  Two aligned 16B LDS reads.
DEV v16h frag_a(const _Float16* row, int hi) {
  v8h lo = *(const v8h*)(row + hi * 8);
  v8h hg = *(const v8h*)(row + 16 + hi * 8);
  v16h f;
#pragma unroll
  for (int i = 0; i < 8; ++i) { f[i] = lo[i]; f[i + 8] = hg[i]; }
  return f;
}

// B fragment (32x16 f16, KxN) read from a K-major (transposed) LDS tile:
// lane = hi*16 + n, element e -> k = (e>>3)*16 + hi*8 + (e&7).
DEV v16h frag_b(const _Float16* rowN, int hi) {
  v8h lo = *(const v8h*)(rowN + hi * 8);
  v8h hg = *(const v8h*)(rowN + 16 + hi * 8);
  v16h f;
#pragma unroll
  for (int i = 0; i < 8; ++i) { f[i] = lo[i]; f[i + 8] = hg[i]; }
  return f;
}

// ---------------------------------------------------------------------------
// Generic 4096x1024x1024 projection GEMM.
//   A_F16 = false : A is fp32 (q/k/v activations), converted to f16 in staging
//   A_F16 = true  : A is f16 (concat-heads O from workspace, async DMA path)
//   SPLIT = true  : write f16 output in [b][h][s][64] head-split layout
//   SPLIT = false : write fp32 output row-major (final projection)
// Block tile 128x128, 8 waves in a 2(M)x4(N) grid, wave tile 64x32.
// ---------------------------------------------------------------------------
template <bool A_F16, bool SPLIT>
__global__ __launch_bounds__(256) void gemm_proj(
    const void* __restrict__ Ap, const float* __restrict__ W,
    const float* __restrict__ bias, _Float16* __restrict__ dst16,
    float* __restrict__ dst32) {
  constexpr int N = DM, K = DM;
  __shared__ _Float16 sA[128][40];   // [m][k], ld=40 halves (80B, 16B aligned)
  __shared__ _Float16 sB[128][40];   // B transposed: [n][k]

  const int tid = threadIdx.x;
  const int n0 = blockIdx.x * 128;
  const int m0 = blockIdx.y * 128;
  const int lane = tid & 31, w = tid >> 5;
  const int wm = (w >> 2) * 64, wn = (w & 3) * 32;
  const int ln = lane & 15, hi = lane >> 4;

  v8f acc[4][2] = {};

  for (int k0 = 0; k0 < K; k0 += 32) {
    // ---- stage A tile (128x32) ----
    if constexpr (A_F16) {
      const _Float16* A = (const _Float16*)Ap;
#pragma unroll
      for (int i = 0; i < 2; ++i) {
        int idx = tid + i * 256;                 // 512 groups of 8 halves
        int m = idx >> 2, kg = (idx & 3) * 8;
        cp16_g2l(&sA[m][kg], A + (size_t)(m0 + m) * K + k0 + kg);
      }
    } else {
      const float* A = (const float*)Ap;
#pragma unroll
      for (int i = 0; i < 4; ++i) {
        int idx = tid + i * 256;                 // 1024 groups of 4 floats
        int m = idx >> 3, kg = (idx & 7) * 4;
        f32x4 vv = *(const f32x4*)(A + (size_t)(m0 + m) * K + k0 + kg);
        sA[m][kg + 0] = (_Float16)vv[0]; sA[m][kg + 1] = (_Float16)vv[1];
        sA[m][kg + 2] = (_Float16)vv[2]; sA[m][kg + 3] = (_Float16)vv[3];
      }
    }
    // ---- stage B tile transposed: W[k][n] -> sB[n][k] ----
#pragma unroll
    for (int i = 0; i < 4; ++i) {
      int idx = tid + i * 256;                   // 32 k-rows * 32 n-groups
      int k = idx >> 5, ng = (idx & 31) * 4;
      f32x4 vv = *(const f32x4*)(W + (size_t)(k0 + k) * N + n0 + ng);
      sB[ng + 0][k] = (_Float16)vv[0]; sB[ng + 1][k] = (_Float16)vv[1];
      sB[ng + 2][k] = (_Float16)vv[2]; sB[ng + 3][k] = (_Float16)vv[3];
    }
    if (k0 + 32 < K)                              // gfx1250 global_prefetch_b8
      __builtin_prefetch(W + (size_t)(k0 + 32) * N + n0 + (tid & 31) * 4, 0, 1);
    if constexpr (A_F16) async_copy_fence();
    __syncthreads();

    v16h a[4], b[2];
#pragma unroll
    for (int mi = 0; mi < 4; ++mi) a[mi] = frag_a(&sA[wm + mi * 16 + ln][0], hi);
#pragma unroll
    for (int ni = 0; ni < 2; ++ni) b[ni] = frag_b(&sB[wn + ni * 16 + ln][0], hi);
#pragma unroll
    for (int mi = 0; mi < 4; ++mi)
#pragma unroll
      for (int ni = 0; ni < 2; ++ni)
        acc[mi][ni] = wmma16(a[mi], b[ni], acc[mi][ni]);
    __syncthreads();
  }

  // ---- epilogue: D layout lane=hi*16+n, vgpr r -> row hi*8+r ----
#pragma unroll
  for (int mi = 0; mi < 4; ++mi)
#pragma unroll
    for (int ni = 0; ni < 2; ++ni)
#pragma unroll
      for (int r = 0; r < 8; ++r) {
        int mg = m0 + wm + mi * 16 + hi * 8 + r;
        int ng = n0 + wn + ni * 16 + ln;
        float v = acc[mi][ni][r] + bias[ng];
        if constexpr (SPLIT) {
          int bb = mg >> 11, s = mg & (SEQ - 1);
          int h = ng >> 6, d = ng & (DH - 1);
          dst16[(((size_t)(bb * NH + h) * SEQ) + s) * DH + d] = (_Float16)v;
        } else {
          dst32[(size_t)mg * DM + ng] = v;
        }
      }
}

// ---------------------------------------------------------------------------
// logits[b,h,q,n] = (Qh . Kh) * 1/sqrt(64) + mask[b,n] * -1e9
// Per (b,h): M=2048, N=2048, K=64. Block tile 128x128, single LDS stage
// fed by the async global->LDS DMA (pure f16 copies, no conversion).
// B operand is Kh^T, i.e. K-major tile == Kh row-major: copy directly.
// ---------------------------------------------------------------------------
__global__ __launch_bounds__(256) void attn_logits(
    const _Float16* __restrict__ qh, const _Float16* __restrict__ kh,
    const float* __restrict__ mask, float* __restrict__ attn) {
  __shared__ _Float16 sQ[128][72];   // [q][k] ld=72 (144B rows, 16B aligned)
  __shared__ _Float16 sK[128][72];   // [n][k]
  const int tid = threadIdx.x;
  const int q0 = blockIdx.x * 128, n0 = blockIdx.y * 128;
  const int bh = blockIdx.z, bb = bh >> 4;
  const _Float16* Q = qh + ((size_t)bh * SEQ + q0) * DH;
  const _Float16* Kb = kh + ((size_t)bh * SEQ + n0) * DH;

#pragma unroll
  for (int i = 0; i < 4; ++i) {
    int idx = tid + i * 256;                     // 1024 groups of 8 halves
    int r = idx >> 3, cg = (idx & 7) * 8;
    cp16_g2l(&sQ[r][cg], Q + (size_t)r * DH + cg);
    cp16_g2l(&sK[r][cg], Kb + (size_t)r * DH + cg);
  }
  async_copy_fence();
  __syncthreads();

  const int lane = tid & 31, w = tid >> 5;
  const int wm = (w >> 2) * 64, wn = (w & 3) * 32;
  const int ln = lane & 15, hi = lane >> 4;

  v8f acc[4][2] = {};
#pragma unroll
  for (int ks = 0; ks < DH; ks += 32) {
    v16h a[4], b[2];
#pragma unroll
    for (int mi = 0; mi < 4; ++mi) a[mi] = frag_a(&sQ[wm + mi * 16 + ln][ks], hi);
#pragma unroll
    for (int ni = 0; ni < 2; ++ni) b[ni] = frag_b(&sK[wn + ni * 16 + ln][ks], hi);
#pragma unroll
    for (int mi = 0; mi < 4; ++mi)
#pragma unroll
      for (int ni = 0; ni < 2; ++ni)
        acc[mi][ni] = wmma16(a[mi], b[ni], acc[mi][ni]);
  }

  float* outp = attn + (size_t)bh * SEQ * SEQ;
#pragma unroll
  for (int mi = 0; mi < 4; ++mi)
#pragma unroll
    for (int ni = 0; ni < 2; ++ni)
#pragma unroll
      for (int r = 0; r < 8; ++r) {
        int qg = q0 + wm + mi * 16 + hi * 8 + r;
        int ng = n0 + wn + ni * 16 + ln;
        float v = acc[mi][ni][r] * 0.125f + mask[bb * SEQ + ng] * -1e9f;
        outp[(size_t)qg * SEQ + ng] = v;
      }
}

// ---------------------------------------------------------------------------
// In-place row softmax over attn, one block per row of 2048.
// ---------------------------------------------------------------------------
__global__ __launch_bounds__(256) void softmax_rows(float* __restrict__ attn) {
  const size_t row = (size_t)blockIdx.y * gridDim.x + blockIdx.x;
  float* p = attn + row * SEQ;
  const int tid = threadIdx.x;
  __shared__ float red[256];

  float x[8];
  float m = -3.4e38f;
#pragma unroll
  for (int j = 0; j < 8; ++j) { x[j] = p[tid + j * 256]; m = fmaxf(m, x[j]); }
  red[tid] = m; __syncthreads();
  for (int s = 128; s > 0; s >>= 1) {
    if (tid < s) red[tid] = fmaxf(red[tid], red[tid + s]);
    __syncthreads();
  }
  m = red[0]; __syncthreads();

  float sum = 0.f;
#pragma unroll
  for (int j = 0; j < 8; ++j) { x[j] = __expf(x[j] - m); sum += x[j]; }
  red[tid] = sum; __syncthreads();
  for (int s = 128; s > 0; s >>= 1) {
    if (tid < s) red[tid] += red[tid + s];
    __syncthreads();
  }
  float inv = 1.0f / red[0];
#pragma unroll
  for (int j = 0; j < 8; ++j) p[tid + j * 256] = x[j] * inv;
}

// ---------------------------------------------------------------------------
// O = attn @ Vh per (b,h): M=2048, N=64, K=2048. Block tile 128x64,
// 8 waves in 4(M)x2(N) grid, wave tile 32x32. Output concat-heads f16.
// ---------------------------------------------------------------------------
__global__ __launch_bounds__(256) void attn_av(
    const float* __restrict__ attn, const _Float16* __restrict__ vh,
    _Float16* __restrict__ oc) {
  __shared__ _Float16 sA[128][40];   // attn tile [q][k], f32 -> f16
  __shared__ _Float16 sV[64][40];    // V transposed: [d][k]
  const int tid = threadIdx.x;
  const int q0 = blockIdx.x * 128;
  const int bh = blockIdx.y, bb = bh >> 4, h = bh & 15;
  const float* Ab = attn + ((size_t)bh * SEQ + q0) * SEQ;
  const _Float16* Vb = vh + (size_t)bh * SEQ * DH;

  const int lane = tid & 31, w = tid >> 5;
  const int wm = (w & 3) * 32, wn = (w >> 2) * 32;
  const int ln = lane & 15, hi = lane >> 4;

  v8f acc[2][2] = {};
  for (int k0 = 0; k0 < SEQ; k0 += 32) {
#pragma unroll
    for (int i = 0; i < 4; ++i) {
      int idx = tid + i * 256;                   // 128 rows * 8 float4 groups
      int r = idx >> 3, cg = (idx & 7) * 4;
      f32x4 vv = *(const f32x4*)(Ab + (size_t)r * SEQ + k0 + cg);
      sA[r][cg + 0] = (_Float16)vv[0]; sA[r][cg + 1] = (_Float16)vv[1];
      sA[r][cg + 2] = (_Float16)vv[2]; sA[r][cg + 3] = (_Float16)vv[3];
    }
    {                                            // Vb[k][d] -> sV[d][k]
      int k = tid >> 3, dg = (tid & 7) * 8;
      v8h vv = *(const v8h*)(Vb + (size_t)(k0 + k) * DH + dg);
#pragma unroll
      for (int j = 0; j < 8; ++j) sV[dg + j][k] = vv[j];
    }
    __syncthreads();

    v16h a[2], b[2];
#pragma unroll
    for (int mi = 0; mi < 2; ++mi) a[mi] = frag_a(&sA[wm + mi * 16 + ln][0], hi);
#pragma unroll
    for (int ni = 0; ni < 2; ++ni) b[ni] = frag_b(&sV[wn + ni * 16 + ln][0], hi);
#pragma unroll
    for (int mi = 0; mi < 2; ++mi)
#pragma unroll
      for (int ni = 0; ni < 2; ++ni)
        acc[mi][ni] = wmma16(a[mi], b[ni], acc[mi][ni]);
    __syncthreads();
  }

#pragma unroll
  for (int mi = 0; mi < 2; ++mi)
#pragma unroll
    for (int ni = 0; ni < 2; ++ni)
#pragma unroll
      for (int r = 0; r < 8; ++r) {
        int qg = q0 + wm + mi * 16 + hi * 8 + r;
        int d  = wn + ni * 16 + ln;
        oc[((size_t)(bb * SEQ + qg)) * DM + h * DH + d] = (_Float16)acc[mi][ni][r];
      }
}

// ---------------------------------------------------------------------------
extern "C" void kernel_launch(void* const* d_in, const int* in_sizes, int n_in,
                              void* d_out, int out_size, void* d_ws,
                              size_t ws_size, hipStream_t stream) {
  (void)in_sizes; (void)n_in; (void)out_size; (void)ws_size;

  const float* q    = (const float*)d_in[0];
  const float* k    = (const float*)d_in[1];
  const float* v    = (const float*)d_in[2];
  const float* mask = (const float*)d_in[3];
  const float* wq   = (const float*)d_in[4];
  const float* bq   = (const float*)d_in[5];
  const float* wk   = (const float*)d_in[6];
  const float* bk   = (const float*)d_in[7];
  const float* wv   = (const float*)d_in[8];
  const float* bv   = (const float*)d_in[9];
  const float* wo   = (const float*)d_in[10];
  const float* bo   = (const float*)d_in[11];

  float* out  = (float*)d_out;
  float* attn = out + (size_t)MROWS * DM;        // attn after out, flat

  _Float16* qh = (_Float16*)d_ws;                // [b][h][s][64]
  _Float16* kh = qh + (size_t)MROWS * DM;
  _Float16* vh = kh + (size_t)MROWS * DM;
  _Float16* oc = vh + (size_t)MROWS * DM;        // [b][s][1024]

  const dim3 gproj(DM / 128, MROWS / 128);       // (8, 32)

  gemm_proj<false, true><<<gproj, 256, 0, stream>>>(q, wq, bq, qh, nullptr);
  gemm_proj<false, true><<<gproj, 256, 0, stream>>>(k, wk, bk, kh, nullptr);
  gemm_proj<false, true><<<gproj, 256, 0, stream>>>(v, wv, bv, vh, nullptr);

  attn_logits<<<dim3(SEQ / 128, SEQ / 128, 2 * NH), 256, 0, stream>>>(
      qh, kh, mask, attn);

  softmax_rows<<<dim3(SEQ, 2 * NH), 256, 0, stream>>>(attn);

  attn_av<<<dim3(SEQ / 128, 2 * NH), 256, 0, stream>>>(attn, vh, oc);

  gemm_proj<true, false><<<gproj, 256, 0, stream>>>(oc, wo, bo, nullptr, out);
}
